// Multi_stage_VQ_54056458387898
// MI455X (gfx1250) — matrix operand order repaired and checked by reading, not compiled, not run on hardware
//
#include <hip/hip_runtime.h>
#include <math.h>

// ---------------------------------------------------------------------------
// Multi-stage VQ for MI455X (gfx1250, wave32).
// v3: same as v2 but with the 6-argument tensor_load_to_lds builtin
// (amdgpu-toolchain / clang-23 arity): codebook tiles staged into LDS by the
// Tensor Data Mover (double buffered; TDM pad feature produces the
// bank-skewed 528B stride), WMMAs feed from LDS on both operands; register
// argmin + fused epilogue unchanged.
// ---------------------------------------------------------------------------

typedef __attribute__((ext_vector_type(16))) __bf16 v16bf;
typedef __attribute__((ext_vector_type(8)))  float  v8f;
typedef __attribute__((ext_vector_type(4)))  unsigned int v4u;
typedef __attribute__((ext_vector_type(8)))  int v8i;
typedef __attribute__((ext_vector_type(4)))  int v4i;

union V16U { struct { uint4 lo, hi; } u; v16bf v; };

__device__ __forceinline__ unsigned short f2bf(float f) {
  // round-to-nearest-even float -> bf16 (bit trick)
  unsigned int u = __float_as_uint(f);
  u += 0x7FFFu + ((u >> 16) & 1u);
  return (unsigned short)(u >> 16);
}

#define VQ_N 65536     // 16*4096 flattened rows
#define VQ_D 256
#define VQ_K 512
#define BROW 264       // padded LDS row stride in ushorts (528 B, 16B aligned)

// Issue one TDM copy: 64 codebook rows (one 64-column score group) = 32KB
// contiguous in global, written to LDS with 16B padding every 512B row
// (pad_interval=128 dwords -> code 6, pad_amount=4 dwords -> code 3).
__device__ __forceinline__ void tdm_load_b_tile(const unsigned short* gsrc,
                                                unsigned int ldsAddr) {
  const unsigned long long ga = (unsigned long long)(uintptr_t)gsrc;
  v4u g0 = { 1u,                                   // count=1 (valid descriptor)
             ldsAddr,                              // lds_addr
             (unsigned int)ga,                     // global_addr[31:0]
             (unsigned int)(ga >> 32) | 0x80000000u }; // [56:32] | type=2<<30
  v8i g1 = { (int)((2u << 16)       // data_size = 4B
                 | (1u << 20)       // pad_enable
                 | (6u << 22)       // pad_interval: 128 dwords
                 | (3u << 25)),     // pad_amount: 4 dwords
             (int)(128u << 16),     // atomic_barrier_addr=0 | tensor_dim0.lo=128
             (int)(64u << 16),      // tensor_dim0.hi=0 | tensor_dim1.lo=64
             (int)(128u << 16),     // tensor_dim1.hi=0 | tile_dim0=128
             64,                    // tile_dim1=64 | tile_dim2=0
             128,                   // tensor_dim0_stride.lo=128
             0, 0 };                // stride0.hi | tensor_dim1_stride=0
  v4i gz4 = { 0, 0, 0, 0 };         // groups 2/3 unused (2-D tensor)
  v8i gz8 = { 0, 0, 0, 0, 0, 0, 0, 0 };
  __builtin_amdgcn_tensor_load_to_lds(g0, g1, gz4, gz4, gz8, 0);
}

// ---------------------------------------------------------------------------
// Zero counters / loss partials
// ---------------------------------------------------------------------------
__global__ void vq_zero_kernel(int* __restrict__ counts, float* __restrict__ sums) {
  int t = blockIdx.x * blockDim.x + threadIdx.x;
  if (t < 3 * VQ_K) counts[t] = 0;
  if (t < 4) sums[t] = 0.0f;
}

// ---------------------------------------------------------------------------
// Codebook prep: f32 -> bf16 copies + per-row squared norms (f32).
// ---------------------------------------------------------------------------
__global__ __launch_bounds__(128)
void vq_prep_kernel(const float* __restrict__ cb0, const float* __restrict__ cb1,
                    const float* __restrict__ cb2,
                    unsigned short* __restrict__ cbbf, float* __restrict__ norms) {
  const int gw   = (blockIdx.x * 128 + threadIdx.x) >> 5;   // global wave id
  const int lane = threadIdx.x & 31;
  if (gw >= 3 * VQ_K) return;
  const int s = gw >> 9;
  const int k = gw & (VQ_K - 1);
  const float* src = (s == 0 ? cb0 : (s == 1 ? cb1 : cb2)) + (size_t)k * VQ_D;
  unsigned short* dst = cbbf + (size_t)gw * VQ_D;
  float ss = 0.0f;
#pragma unroll
  for (int j = 0; j < 2; ++j) {
    const int c = lane * 8 + j * 4;
    float4 v = *(const float4*)(src + c);
    ss += v.x * v.x + v.y * v.y + v.z * v.z + v.w * v.w;
    unsigned int lo = (unsigned int)f2bf(v.x) | ((unsigned int)f2bf(v.y) << 16);
    unsigned int hi = (unsigned int)f2bf(v.z) | ((unsigned int)f2bf(v.w) << 16);
    *(uint2*)(dst + c) = make_uint2(lo, hi);
  }
#pragma unroll
  for (int off = 16; off >= 1; off >>= 1) ss += __shfl_xor(ss, off, 32);
  if (lane == 0) norms[gw] = ss;
}

// ---------------------------------------------------------------------------
// Fused stage kernel. Block = 128 threads (4 waves), 64 rows/block.
// ---------------------------------------------------------------------------
__global__ __launch_bounds__(128)
void vq_stage_kernel(const float* __restrict__ data,
                     const unsigned short* __restrict__ cbbf,   // [512*256] bf16
                     const float* __restrict__ cbf,             // [512*256] f32
                     const float* __restrict__ norms,           // [512]
                     float* __restrict__ quant,                 // [N*256] in/out (d_out)
                     float* __restrict__ enc,                   // [N*512] one-hot out
                     int* __restrict__ counts,                  // [512]
                     float* __restrict__ loss_sum,              // scalar accumulator
                     int stage) {
  __shared__ unsigned short Ash[64 * BROW];        // residual bf16, 64 rows
  __shared__ unsigned short Bsh[2][64 * BROW];     // codebook tile double buffer
  __shared__ float normsh[VQ_K];

  const int tid      = threadIdx.x;
  const int blockRow = blockIdx.x * 64;
  const int w        = tid >> 5;
  const int lane     = tid & 31;
  const int half     = lane >> 4;
  const int l15      = lane & 15;
  const int wrow     = w * 16;

  // kick off TDM for column group 0 while we stage A
  if (w == 0) tdm_load_b_tile(cbbf, (unsigned int)(uintptr_t)&Bsh[0][0]);

  // ---- stage residual (f32) -> bf16 into LDS ----
  for (int slot = tid; slot < 4096; slot += 128) {        // 64 rows * 64 float4
    const int row = slot >> 6;
    const int c4  = (slot & 63) * 4;
    const size_t g = (size_t)(blockRow + row) * VQ_D + c4;
    float4 d = *(const float4*)(data + g);
    if (stage != 0) {
      const float4 qp = *(const float4*)(quant + g);
      d.x -= qp.x; d.y -= qp.y; d.z -= qp.z; d.w -= qp.w;
    }
    unsigned int lo = (unsigned int)f2bf(d.x) | ((unsigned int)f2bf(d.y) << 16);
    unsigned int hi = (unsigned int)f2bf(d.z) | ((unsigned int)f2bf(d.w) << 16);
    *(uint2*)&Ash[row * BROW + c4] = make_uint2(lo, hi);
  }
  for (int k = tid; k < VQ_K; k += 128) normsh[k] = norms[k];

  if (w == 0) __builtin_amdgcn_s_wait_tensorcnt(0);
  __syncthreads();

  float bestv[8];
  int   besti[8];
#pragma unroll
  for (int i = 0; i < 8; ++i) { bestv[i] = 3.4e38f; besti[i] = 0; }

  for (int cg = 0; cg < 8; ++cg) {          // 8 column groups of 64
    const int buf = cg & 1;
    // prefetch next tile via TDM into the other buffer (safe: barrier ending
    // cg-1 closed all reads of that buffer)
    if (w == 0 && cg + 1 < 8)
      tdm_load_b_tile(cbbf + (size_t)(cg + 1) * 64 * VQ_D,
                      (unsigned int)(uintptr_t)&Bsh[buf ^ 1][0]);

    v8f acc[4];
#pragma unroll
    for (int t = 0; t < 4; ++t)
#pragma unroll
      for (int j = 0; j < 8; ++j) acc[t][j] = 0.0f;

    for (int dc = 0; dc < 8; ++dc) {        // K = 256 in chunks of 32
      // A operand (16x32 bf16): lane l15 = row M; half selects K sub-ranges
      const unsigned short* ap = &Ash[(wrow + l15) * BROW + dc * 32 + half * 8];
      V16U a;
      a.u.lo = *(const uint4*)ap;           // K = half*8 .. +8
      a.u.hi = *(const uint4*)(ap + 16);    // K = 16+half*8 .. +8
#pragma unroll
      for (int t = 0; t < 4; ++t) {
        // B operand (32x16 bf16) from LDS: lane l15 = col N; K = half*8 .. +8, 16+half*8 .. +8
        const unsigned short* bp =
            &Bsh[buf][(t * 16 + l15) * BROW + dc * 32 + half * 8];
        V16U b;
        b.u.lo = *(const uint4*)bp;
        b.u.hi = *(const uint4*)(bp + 16);
        acc[t] = __builtin_amdgcn_wmma_f32_16x16x32_bf16(
            false, a.v, false, b.v, (short)0, acc[t], false, false);
      }
    }
    // fold this column group into the running argmin (d = ||cb||^2 - 2*score)
#pragma unroll
    for (int t = 0; t < 4; ++t) {
      const int col = cg * 64 + t * 16 + l15;
      const float nk = normsh[col];
#pragma unroll
      for (int i = 0; i < 8; ++i) {
        const float dv = fmaf(-2.0f, acc[t][i], nk);
        if (dv < bestv[i]) { bestv[i] = dv; besti[i] = col; }
      }
    }
    if (cg + 1 < 8) {
      if (w == 0) __builtin_amdgcn_s_wait_tensorcnt(0);
      __syncthreads();                      // publish next B tile
    }
  }

  // cross-lane argmin within each 16-lane half (each half owns 8 rows);
  // tie-break on lower index to match jnp.argmin first-match semantics
#pragma unroll
  for (int off = 1; off < 16; off <<= 1) {
#pragma unroll
    for (int i = 0; i < 8; ++i) {
      const float ov = __shfl_xor(bestv[i], off, 32);
      const int   oi = __shfl_xor(besti[i], off, 32);
      if (ov < bestv[i] || (ov == bestv[i] && oi < besti[i])) {
        bestv[i] = ov; besti[i] = oi;
      }
    }
  }

  // ---- fused epilogue: one-hot, gather, quantized +=, MSE, histogram ----
  const int growBase = blockRow + wrow + half * 8;
  float errsum = 0.0f;
#pragma unroll 1
  for (int i = 0; i < 8; ++i) {
    const int row = growBase + i;
    const int idx = besti[i];               // uniform across the 16-lane half

    float* encrow = enc + (size_t)row * VQ_K;
#pragma unroll
    for (int j = 0; j < 8; ++j) {
      const int c0 = l15 * 32 + j * 4;
      float4 ev;
      ev.x = (c0 + 0 == idx) ? 1.0f : 0.0f;
      ev.y = (c0 + 1 == idx) ? 1.0f : 0.0f;
      ev.z = (c0 + 2 == idx) ? 1.0f : 0.0f;
      ev.w = (c0 + 3 == idx) ? 1.0f : 0.0f;
      *(float4*)(encrow + c0) = ev;
    }

    const float* qrow = cbf  + (size_t)idx * VQ_D;   // f32 gather (exact)
    const float* drow = data + (size_t)row * VQ_D;
    float*       prow = quant + (size_t)row * VQ_D;
#pragma unroll
    for (int j = 0; j < 4; ++j) {
      const int d0 = l15 * 16 + j * 4;
      const float4 q  = *(const float4*)(qrow + d0);
      const float4 dv = *(const float4*)(drow + d0);
      float4 qp;
      if (stage != 0) qp = *(const float4*)(prow + d0);
      else { qp.x = 0.f; qp.y = 0.f; qp.z = 0.f; qp.w = 0.f; }
      const float ex = q.x - (dv.x - qp.x);
      const float ey = q.y - (dv.y - qp.y);
      const float ez = q.z - (dv.z - qp.z);
      const float ew = q.w - (dv.w - qp.w);
      errsum += ex * ex + ey * ey + ez * ez + ew * ew;
      float4 o;
      o.x = qp.x + q.x; o.y = qp.y + q.y; o.z = qp.z + q.z; o.w = qp.w + q.w;
      *(float4*)(prow + d0) = o;
    }
    if (l15 == 0) atomicAdd(&counts[idx], 1);       // lanes 0 & 16: one per half
  }
#pragma unroll
  for (int off = 16; off >= 1; off >>= 1) errsum += __shfl_xor(errsum, off, 32);
  if (lane == 0) atomicAdd(loss_sum, errsum);
}

// ---------------------------------------------------------------------------
// Finalize: total_loss = 1.25 * sum(sq)/(N*D); entropy = -sum p log(p+eps)
// ---------------------------------------------------------------------------
__global__ __launch_bounds__(512)
void vq_finalize_kernel(const int* __restrict__ counts, const float* __restrict__ sums,
                        float* __restrict__ out2) {
  __shared__ float sent;
  const int t = threadIdx.x;
  if (t == 0) sent = 0.0f;
  __syncthreads();
  float e = 0.0f;
  for (int s = 0; s < 3; ++s) {
    const float p = (float)counts[s * VQ_K + t] * (1.0f / (float)VQ_N);
    e -= p * logf(p + 1e-10f);
  }
  atomicAdd(&sent, e);
  __syncthreads();
  if (t == 0) {
    out2[0] = 1.25f * (sums[0] + sums[1] + sums[2]) * (1.0f / ((float)VQ_N * (float)VQ_D));
    out2[1] = sent;
  }
}

// ---------------------------------------------------------------------------
extern "C" void kernel_launch(void* const* d_in, const int* in_sizes, int n_in,
                              void* d_out, int out_size, void* d_ws, size_t ws_size,
                              hipStream_t stream) {
  const float* data = (const float*)d_in[0];
  const float* cb[3] = {(const float*)d_in[1], (const float*)d_in[2], (const float*)d_in[3]};
  // d_in[4] = `using`; setup uses 0 => all 3 stages (the only case where the
  // reference returns all three encodings), so it is intentionally ignored.
  (void)in_sizes; (void)n_in; (void)out_size; (void)ws_size;

  float* out = (float*)d_out;
  float* quant = out;                                          // 16777216 f32
  const size_t QOFF = (size_t)VQ_N * VQ_D;                     // 16777216
  const size_t EN   = (size_t)VQ_N * VQ_K;                     // 33554432
  float* scal = out + QOFF + 3 * EN;                           // loss, entropy
  float* encs[3] = { out + QOFF, out + QOFF + EN, out + QOFF + 2 * EN };

  char* ws = (char*)d_ws;
  unsigned short* cbbf = (unsigned short*)ws;                         // 786432 B
  float* norms = (float*)(ws + 3 * VQ_K * VQ_D * sizeof(unsigned short));
  int*   counts = (int*)((char*)norms + 3 * VQ_K * sizeof(float));
  float* sums   = (float*)((char*)counts + 3 * VQ_K * sizeof(int));

  vq_zero_kernel<<<6, 256, 0, stream>>>(counts, sums);
  vq_prep_kernel<<<(3 * VQ_K * 32) / 128, 128, 0, stream>>>(cb[0], cb[1], cb[2], cbbf, norms);
  for (int s = 0; s < 3; ++s) {
    vq_stage_kernel<<<VQ_N / 64, 128, 0, stream>>>(
        data, cbbf + (size_t)s * VQ_K * VQ_D, cb[s], norms + s * VQ_K,
        quant, encs[s], counts + s * VQ_K, sums + s, s);
  }
  vq_finalize_kernel<<<1, 512, 0, stream>>>(counts, sums, scal);
}
